// LSTM2_69793218560704
// MI455X (gfx1250) — 8-bit matrix operand order repaired and checked by reading, compiled, not run on hardware
//
#include <hip/hip_runtime.h>
#include <cstdint>

#define BATCH 16384
#define TT    128
#define FF    9
#define HH    128
#define OO    12

#define LDS_BHH0   0
#define LDS_BIH1   65536
#define LDS_BHH1   131072
#define LDS_BEFF   196608
#define LDS_BIAS0  229376
#define LDS_BIAS1  231424
#define LDS_STAGE  233472
#define STAGE_PER_WAVE 4608
#define LDS_TOTAL  (233472 + 8 * STAGE_PER_WAVE)

typedef __attribute__((ext_vector_type(8)))  int   v8i;
typedef __attribute__((ext_vector_type(16))) int   v16i;
typedef __attribute__((ext_vector_type(8)))  float v8f;

__device__ __forceinline__ unsigned pk4_fp8(float a, float b, float c, float d) {
    int r = 0;
    r = __builtin_amdgcn_cvt_pk_fp8_f32(a, b, r, false);
    r = __builtin_amdgcn_cvt_pk_fp8_f32(c, d, r, true);
    return (unsigned)r;
}

#if defined(__has_builtin)
#if __has_builtin(__builtin_amdgcn_tanh_f32)
#define HAVE_HW_TANH 1
#endif
#endif

__device__ __forceinline__ float tanh_(float x) {
#ifdef HAVE_HW_TANH
    return __builtin_amdgcn_tanh_f32(x);
#else
    return 2.0f / (1.0f + __expf(-2.0f * x)) - 1.0f;
#endif
}
__device__ __forceinline__ float sigf(float x) {
#ifdef HAVE_HW_TANH
    return fmaf(0.5f, __builtin_amdgcn_tanh_f32(0.5f * x), 0.5f);
#else
    return 1.0f / (1.0f + __expf(-x));
#endif
}

__device__ __forceinline__ v16i ld_frag128(const char* p) {
    const int4* q = (const int4*)p;
    int4 a = q[0], b = q[1], c = q[2], d = q[3];
    v16i r;
    r[0]=a.x;  r[1]=a.y;  r[2]=a.z;  r[3]=a.w;
    r[4]=b.x;  r[5]=b.y;  r[6]=b.z;  r[7]=b.w;
    r[8]=c.x;  r[9]=c.y;  r[10]=c.z; r[11]=c.w;
    r[12]=d.x; r[13]=d.y; r[14]=d.z; r[15]=d.w;
    return r;
}
__device__ __forceinline__ v8i ld_frag64(const char* p) {
    const int4* q = (const int4*)p;
    int4 a = q[0], b = q[1];
    v8i r;
    r[0]=a.x; r[1]=a.y; r[2]=a.z; r[3]=a.w;
    r[4]=b.x; r[5]=b.y; r[6]=b.z; r[7]=b.w;
    return r;
}
__device__ __forceinline__ v8f wmma64(v8i a, v8i b, v8f c) {
    return __builtin_amdgcn_wmma_f32_16x16x64_fp8_fp8(a, b, (short)0, c, false, false);
}
__device__ __forceinline__ v8f wmma128(v16i a, v16i b, v8f c) {
    return __builtin_amdgcn_wmma_f32_16x16x128_fp8_fp8(a, b, (short)0, c, false, false);
}

__global__ __launch_bounds__(256, 1)
void lstm2_scan_kernel(const float* __restrict__ x,
                       const float* __restrict__ W_in, const float* __restrict__ b_in,
                       const float* __restrict__ Wih0, const float* __restrict__ Whh0,
                       const float* __restrict__ bih0, const float* __restrict__ bhh0,
                       const float* __restrict__ Wih1, const float* __restrict__ Whh1,
                       const float* __restrict__ bih1, const float* __restrict__ bhh1,
                       float* __restrict__ h1_out  ) {
    extern __shared__ char smem[];
    const int tid  = threadIdx.x;
    const int lane = tid & 31;
    const int wave = tid >> 5;
    const int col  = lane & 15;
    const int hi   = lane >> 4;

    {
        const float* mats[3] = {Whh0, Wih1, Whh1};
        char* dsts[3] = {smem + LDS_BHH0, smem + LDS_BIH1, smem + LDS_BHH1};
        for (int m = 0; m < 3; ++m) {
            const float* W = mats[m];
            unsigned* dst = (unsigned*)dsts[m];
            for (int idx = tid; idx < 512 * 32; idx += 256) {
                int n  = idx >> 5;
                int k4 = (idx & 31) << 2;
                float4 w = *(const float4*)(W + n * 128 + k4);
                unsigned d = pk4_fp8(w.x, w.y, w.z, w.w);
                int frag = n >> 4, c = n & 15;
                int h2 = (k4 >> 4) & 1;
                int v  = ((k4 >> 5) << 2) + ((k4 & 15) >> 2);
                dst[frag * 512 + (h2 * 16 + c) * 16 + v] = d;
            }
        }
        unsigned* be = (unsigned*)(smem + LDS_BEFF);
        for (int i = tid; i < 32768 / 4; i += 256) be[i] = 0u;
    }
    __syncthreads();

    {
        unsigned* be = (unsigned*)(smem + LDS_BEFF);
        for (int idx = tid; idx < 512 * 3; idx += 256) {
            int n  = idx / 3;
            int k4 = (idx % 3) * 4;
            float f[4] = {0.f, 0.f, 0.f, 0.f};
            for (int kk = 0; kk < 4; ++kk) {
                int k = k4 + kk;
                if (k < FF) {
                    float s = 0.f;
                    for (int h = 0; h < 128; ++h) s += Wih0[n * 128 + h] * W_in[h * FF + k];
                    f[kk] = s;
                }
            }
            unsigned d = pk4_fp8(f[0], f[1], f[2], f[3]);
            int frag = n >> 4, c = n & 15;
            int v = k4 >> 2;
            be[frag * 256 + c * 8 + v] = d;
        }
        float* bz0 = (float*)(smem + LDS_BIAS0);
        float* bz1 = (float*)(smem + LDS_BIAS1);
        for (int n = tid; n < 512; n += 256) {
            float s = bih0[n] + bhh0[n];
            for (int h = 0; h < 128; ++h) s += b_in[h] * Wih0[n * 128 + h];
            bz0[n] = s;
            bz1[n] = bih1[n] + bhh1[n];
        }
    }
    __syncthreads();

    const int waveRow = blockIdx.x * 256 + wave * 32;
    char* stage = smem + LDS_STAGE + wave * STAGE_PER_WAVE;
    const float* bz0 = (const float*)(smem + LDS_BIAS0);
    const float* bz1 = (const float*)(smem + LDS_BIAS1);

    const v8f zeroC = (v8f){0.f, 0.f, 0.f, 0.f, 0.f, 0.f, 0.f, 0.f};

    v16i h0f[2] = {}, h1f[2] = {};
    float c0[2][8][8], c1[2][8][8];
#pragma unroll
    for (int a = 0; a < 2; ++a)
#pragma unroll
        for (int j = 0; j < 8; ++j)
#pragma unroll
            for (int e = 0; e < 8; ++e) { c0[a][j][e] = 0.f; c1[a][j][e] = 0.f; }

    auto loadAfrag = [&](int a) -> v16i {
        v16i r;
        const int rrow = a * 16 + col;
#pragma unroll
        for (int p = 0; p < 8; ++p) {
            int v  = p * 2;
            int ks = 32 * (v >> 2) + 16 * hi + 4 * (v & 3);
            uint2 q = *(const uint2*)(stage + rrow * 136 + ks);
            r[v] = (int)q.x; r[v + 1] = (int)q.y;
        }
        return r;
    };

    for (int t = 0; t < TT; ++t) {
        v8i xf[2];
#pragma unroll
        for (int a = 0; a < 2; ++a) {
            const float* xr = x + (size_t)(waveRow + a * 16 + col) * (TT * FF) + t * FF;
            float x0 = xr[0], x1 = xr[1], x2 = xr[2], x3 = xr[3], x4 = xr[4];
            float x5 = xr[5], x6 = xr[6], x7 = xr[7], x8 = xr[8];
            unsigned lo = pk4_fp8(x0, x1, x2, x3);
            unsigned md = pk4_fp8(x4, x5, x6, x7);
            unsigned s8 = pk4_fp8(x8, 0.f, 0.f, 0.f);
            v8i xa = {};
            xa[0] = hi ? 0 : (int)lo;
            xa[1] = hi ? 0 : (int)md;
            xa[2] = hi ? 0 : (int)s8;
            xf[a] = xa;
        }

#pragma unroll
        for (int j = 0; j < 8; ++j) {
            v8f acc[4][2];
#pragma unroll
            for (int g = 0; g < 4; ++g) {
                const int n = j + g * 8;
                v8i  bfx = ld_frag64(smem + LDS_BEFF + n * 1024 + lane * 32);
                v16i bhh = ld_frag128(smem + LDS_BHH0 + n * 2048 + lane * 64);
#pragma unroll
                for (int a = 0; a < 2; ++a) {
                    v8f c = wmma64(xf[a], bfx, zeroC);
                    c = wmma128(h0f[a], bhh, c);
                    acc[g][a] = c;
                }
            }
            float bvi = bz0[(j +  0) * 16 + col];
            float bvf = bz0[(j +  8) * 16 + col];
            float bvg = bz0[(j + 16) * 16 + col];
            float bvo = bz0[(j + 24) * 16 + col];
#pragma unroll
            for (int a = 0; a < 2; ++a)
#pragma unroll
                for (int e = 0; e < 8; ++e) {
                    float ig = sigf(acc[0][a][e] + bvi);
                    float fg = sigf(acc[1][a][e] + bvf);
                    float gg = tanh_(acc[2][a][e] + bvg);
                    float og = sigf(acc[3][a][e] + bvo);
                    float cc = fg * c0[a][j][e] + ig * gg;
                    c0[a][j][e] = cc;
                    float hh = og * tanh_(cc);
                    unsigned hb = (unsigned)__builtin_amdgcn_cvt_pk_fp8_f32(hh, 0.f, 0, false) & 0xFFu;
                    *(unsigned char*)(stage + (a * 16 + e + hi * 8) * 136 + j * 16 + col) =
                        (unsigned char)hb;
                }
        }
        asm volatile("s_wait_dscnt 0" ::: "memory");
#pragma unroll
        for (int a = 0; a < 2; ++a) h0f[a] = loadAfrag(a);
        asm volatile("" ::: "memory");

#pragma unroll
        for (int j = 0; j < 8; ++j) {
            v8f acc[4][2];
#pragma unroll
            for (int g = 0; g < 4; ++g) {
                const int n = j + g * 8;
                v16i bih = ld_frag128(smem + LDS_BIH1 + n * 2048 + lane * 64);
                v16i bhh = ld_frag128(smem + LDS_BHH1 + n * 2048 + lane * 64);
#pragma unroll
                for (int a = 0; a < 2; ++a) {
                    v8f c = wmma128(h0f[a], bih, zeroC);
                    c = wmma128(h1f[a], bhh, c);
                    acc[g][a] = c;
                }
            }
            float bvi = bz1[(j +  0) * 16 + col];
            float bvf = bz1[(j +  8) * 16 + col];
            float bvg = bz1[(j + 16) * 16 + col];
            float bvo = bz1[(j + 24) * 16 + col];
#pragma unroll
            for (int a = 0; a < 2; ++a)
#pragma unroll
                for (int e = 0; e < 8; ++e) {
                    float ig = sigf(acc[0][a][e] + bvi);
                    float fg = sigf(acc[1][a][e] + bvf);
                    float gg = tanh_(acc[2][a][e] + bvg);
                    float og = sigf(acc[3][a][e] + bvo);
                    float cc = fg * c1[a][j][e] + ig * gg;
                    c1[a][j][e] = cc;
                    float hh = og * tanh_(cc);
                    unsigned hb = (unsigned)__builtin_amdgcn_cvt_pk_fp8_f32(hh, 0.f, 0, false) & 0xFFu;
                    *(unsigned char*)(stage + (a * 16 + e + hi * 8) * 136 + j * 16 + col) =
                        (unsigned char)hb;
                    if (t == TT - 1) {
                        h1_out[(size_t)(waveRow + a * 16 + e + hi * 8) * HH + j * 16 + col] = hh;
                    }
                }
        }
        asm volatile("s_wait_dscnt 0" ::: "memory");
#pragma unroll
        for (int a = 0; a < 2; ++a) h1f[a] = loadAfrag(a);
        asm volatile("" ::: "memory");
    }
}

__global__ __launch_bounds__(256)
void out_head_kernel(const float* __restrict__ h1, const float* __restrict__ Wout,
                     const float* __restrict__ bout, float* __restrict__ out) {
    __shared__ float w[OO * HH];
    __shared__ float bb[OO];
    for (int i = threadIdx.x; i < OO * HH; i += 256) w[i] = Wout[i];
    if (threadIdx.x < OO) bb[threadIdx.x] = bout[threadIdx.x];
    __syncthreads();

    const int row = blockIdx.x * 256 + threadIdx.x;
    float acc[OO];
#pragma unroll
    for (int o = 0; o < OO; ++o) acc[o] = bb[o];
    const float* hr = h1 + (size_t)row * HH;
    for (int k = 0; k < HH; k += 4) {
        float4 hv = *(const float4*)(hr + k);
#pragma unroll
        for (int o = 0; o < OO; ++o) {
            acc[o] += hv.x * w[o * HH + k]     + hv.y * w[o * HH + k + 1]
                    + hv.z * w[o * HH + k + 2] + hv.w * w[o * HH + k + 3];
        }
    }
    float m = acc[0];
#pragma unroll
    for (int o = 1; o < OO; ++o) m = fmaxf(m, acc[o]);
    float s = 0.f;
#pragma unroll
    for (int o = 0; o < OO; ++o) { acc[o] = __expf(acc[o] - m); s += acc[o]; }
    float inv = 1.0f / s;
#pragma unroll
    for (int o = 0; o < OO; ++o) out[(size_t)row * OO + o] = acc[o] * inv;
}

extern "C" void kernel_launch(void* const* d_in, const int* in_sizes, int n_in,
                              void* d_out, int out_size, void* d_ws, size_t ws_size,
                              hipStream_t stream) {
    (void)in_sizes; (void)n_in; (void)out_size; (void)ws_size;
    const float* x    = (const float*)d_in[0];
    const float* W_in = (const float*)d_in[1];
    const float* b_in = (const float*)d_in[2];
    const float* Wih0 = (const float*)d_in[3];
    const float* Whh0 = (const float*)d_in[4];
    const float* bih0 = (const float*)d_in[5];
    const float* bhh0 = (const float*)d_in[6];
    const float* Wih1 = (const float*)d_in[7];
    const float* Whh1 = (const float*)d_in[8];
    const float* bih1 = (const float*)d_in[9];
    const float* bhh1 = (const float*)d_in[10];
    const float* Wout = (const float*)d_in[11];
    const float* bout = (const float*)d_in[12];
    float* out  = (float*)d_out;
    float* h1ws = (float*)d_ws;

    (void)hipFuncSetAttribute(reinterpret_cast<const void*>(lstm2_scan_kernel),
                              hipFuncAttributeMaxDynamicSharedMemorySize, LDS_TOTAL);

    lstm2_scan_kernel<<<BATCH / 256, 256, LDS_TOTAL, stream>>>(
        x, W_in, b_in, Wih0, Whh0, bih0, bhh0, Wih1, Whh1, bih1, bhh1, h1ws);

    out_head_kernel<<<BATCH / 256, 256, 0, stream>>>(h1ws, Wout, bout, out);
}
